// SingleHeadAttention_44753559225124
// MI455X (gfx1250) — matrix operand order repaired and checked by reading, not compiled, not run on hardware
//
#include <hip/hip_runtime.h>
#include <hip/hip_bf16.h>

typedef __bf16  v16bf __attribute__((ext_vector_type(16)));
typedef __bf16  v8bf  __attribute__((ext_vector_type(8)));
typedef __bf16  v4bf  __attribute__((ext_vector_type(4)));
typedef float   v8f   __attribute__((ext_vector_type(8)));
typedef unsigned int v4u __attribute__((ext_vector_type(4)));
typedef int     v8i   __attribute__((ext_vector_type(8)));
typedef int     v4i   __attribute__((ext_vector_type(4)));

#define B_  4
#define T_  2048
#define C_  1024
#define NQKV (3 * C_)
#define SCALE_Q 0.03125f   // 1/sqrt(1024)

__device__ __forceinline__ v8f wmma_bf16(v16bf a, v16bf b, v8f c) {
  return __builtin_amdgcn_wmma_f32_16x16x32_bf16(
      false, a, false, b, (short)0, c, false, false);
}

// A fragment: 16x32 (MxK) bf16 from row-major [rows x ld] at (row0,k0).
__device__ __forceinline__ v16bf load_frag_a(const __bf16* base, int ld,
                                             int row0, int k0, int lane) {
  int r = lane & 15, hi = lane >> 4;
  const __bf16* p = base + (size_t)(row0 + r) * ld + k0 + hi * 8;
  v8bf lo = *(const v8bf*)p;
  v8bf up = *(const v8bf*)(p + 16);
  v16bf out;
#pragma unroll
  for (int i = 0; i < 8; ++i) { out[i] = lo[i]; out[i + 8] = up[i]; }
  return out;
}

// B fragment: 32x16 (KxN) operand from an N-major matrix (A * Bstored^T).
__device__ __forceinline__ v16bf load_frag_b(const __bf16* base, int ld,
                                             int n0, int k0, int lane) {
  int n = lane & 15, hi = lane >> 4;
  const __bf16* p = base + (size_t)(n0 + n) * ld + k0 + hi * 8;
  v8bf a = *(const v8bf*)p;
  v8bf b = *(const v8bf*)(p + 16);
  v16bf out;
#pragma unroll
  for (int i = 0; i < 8; ++i) { out[i] = a[i]; out[i + 8] = b[i]; }
  return out;
}

// ---------------------------------------------------------------------------
// Kernel 1: fp32 -> bf16, 4-wide vectorized grid-stride
// ---------------------------------------------------------------------------
__global__ __launch_bounds__(256) void cvt_bf16(const float* __restrict__ in,
                                                __bf16* __restrict__ out,
                                                size_t n4) {
  size_t i = (size_t)blockIdx.x * blockDim.x + threadIdx.x;
  size_t stride = (size_t)gridDim.x * blockDim.x;
  const float4* in4 = (const float4*)in;
  v4bf* out4 = (v4bf*)out;
  for (; i < n4; i += stride) {
    float4 f = in4[i];
    v4bf o; o[0] = (__bf16)f.x; o[1] = (__bf16)f.y;
    o[2] = (__bf16)f.z; o[3] = (__bf16)f.w;
    out4[i] = o;
  }
}

// ---------------------------------------------------------------------------
// Kernel 2: QKV projection.  M=8192, N=3072, K=1024.
// Wave tile 32x64; explicit two-stage ping-pong over K (no rotate moves,
// loads one stage ahead of the WMMAs).
// ---------------------------------------------------------------------------
__global__ __launch_bounds__(256) void qkv_gemm(
    const __bf16* __restrict__ xb, const __bf16* __restrict__ wb,
    __bf16* __restrict__ qb, __bf16* __restrict__ kb,
    __bf16* __restrict__ vt) {
  int wave = threadIdx.x >> 5;
  int lane = threadIdx.x & 31;
  int m0 = blockIdx.x * 256 + wave * 32;
  int n0 = blockIdx.y * 64;

  v8f acc[2][4];
#pragma unroll
  for (int r = 0; r < 2; ++r)
#pragma unroll
    for (int i = 0; i < 4; ++i) acc[r][i] = v8f{};

  v16bf a0[2], b0[4], a1[2], b1[4];
  a0[0] = load_frag_a(xb, C_, m0,      0, lane);
  a0[1] = load_frag_a(xb, C_, m0 + 16, 0, lane);
#pragma unroll
  for (int i = 0; i < 4; ++i) b0[i] = load_frag_b(wb, C_, n0 + 16 * i, 0, lane);

  for (int k0 = 0; k0 < C_ - 64; k0 += 64) {
    a1[0] = load_frag_a(xb, C_, m0,      k0 + 32, lane);
    a1[1] = load_frag_a(xb, C_, m0 + 16, k0 + 32, lane);
#pragma unroll
    for (int i = 0; i < 4; ++i) b1[i] = load_frag_b(wb, C_, n0 + 16 * i, k0 + 32, lane);
#pragma unroll
    for (int r = 0; r < 2; ++r)
#pragma unroll
      for (int i = 0; i < 4; ++i) acc[r][i] = wmma_bf16(a0[r], b0[i], acc[r][i]);
    a0[0] = load_frag_a(xb, C_, m0,      k0 + 64, lane);
    a0[1] = load_frag_a(xb, C_, m0 + 16, k0 + 64, lane);
#pragma unroll
    for (int i = 0; i < 4; ++i) b0[i] = load_frag_b(wb, C_, n0 + 16 * i, k0 + 64, lane);
#pragma unroll
    for (int r = 0; r < 2; ++r)
#pragma unroll
      for (int i = 0; i < 4; ++i) acc[r][i] = wmma_bf16(a1[r], b1[i], acc[r][i]);
  }
  a1[0] = load_frag_a(xb, C_, m0,      C_ - 32, lane);
  a1[1] = load_frag_a(xb, C_, m0 + 16, C_ - 32, lane);
#pragma unroll
  for (int i = 0; i < 4; ++i) b1[i] = load_frag_b(wb, C_, n0 + 16 * i, C_ - 32, lane);
#pragma unroll
  for (int r = 0; r < 2; ++r)
#pragma unroll
    for (int i = 0; i < 4; ++i) acc[r][i] = wmma_bf16(a0[r], b0[i], acc[r][i]);
#pragma unroll
  for (int r = 0; r < 2; ++r)
#pragma unroll
    for (int i = 0; i < 4; ++i) acc[r][i] = wmma_bf16(a1[r], b1[i], acc[r][i]);

  int col = lane & 15, hi = lane >> 4;
#pragma unroll
  for (int rr = 0; rr < 2; ++rr) {
#pragma unroll
    for (int i = 0; i < 4; ++i) {
      int o = n0 + 16 * i + col;
#pragma unroll
      for (int j = 0; j < 8; ++j) {
        int row = m0 + 16 * rr + j + 8 * hi;
        float v = acc[rr][i][j];
        if (o < C_) {
          qb[(size_t)row * C_ + o] = (__bf16)(v * SCALE_Q);
        } else if (o < 2 * C_) {
          kb[(size_t)row * C_ + (o - C_)] = (__bf16)v;
        } else {
          int bb = row >> 11, t = row & (T_ - 1);
          vt[((size_t)bb * C_ + (o - 2 * C_)) * T_ + t] = (__bf16)v;
        }
      }
    }
  }
}

// ---------------------------------------------------------------------------
// Kernel 3: causal flash attention, 32 query rows/block, 8 waves.
// q strip (32x1024 bf16 = 64KB) staged into LDS by the Tensor Data Mover
// (2-D D#: dim0=1024 elems, dim1=32 rows, data_size=2B) issued by wave 0,
// completed with s_wait_tensorcnt + workgroup barrier.  Fallback: coop copy.
// ---------------------------------------------------------------------------
__global__ __launch_bounds__(256) void flash_attn(
    const __bf16* __restrict__ qb, const __bf16* __restrict__ kbm,
    const __bf16* __restrict__ vt, float* __restrict__ out) {
  __shared__ __bf16 s_q[32][C_];       // 64 KB
  __shared__ float  s_s[32][132];
  __shared__ __bf16 s_p[32][128];
  __shared__ float  s_red[32][8];
  __shared__ float  s_m[32], s_l[32], s_r[32];

  int wave = threadIdx.x >> 5;
  int lane = threadIdx.x & 31;
  int bidx = blockIdx.y;
  int q0 = blockIdx.x * 32;

  const __bf16* qrow = qb  + (size_t)bidx * T_ * C_;
  const __bf16* krow = kbm + (size_t)bidx * T_ * C_;
  const __bf16* vrow = vt  + (size_t)bidx * C_ * T_;

  // ---- q -> LDS ----
#if __has_builtin(__builtin_amdgcn_tensor_load_to_lds)
  if (wave == 0) {
    unsigned lds_off = (unsigned)(size_t)(&s_q[0][0]);     // low 32b = LDS byte offset
    unsigned long long ga = (unsigned long long)(const void*)(qrow + (size_t)q0 * C_);
    // D# group0: count=1 | lds_addr | global_addr[56:0] | type=2
    v4u g0 = { 1u, lds_off, (unsigned)(ga & 0xffffffffu),
               (unsigned)((ga >> 32) & 0x01ffffffu) | 0x80000000u };
    // D# group1: data_size=2B; tensor_dim0=1024, tensor_dim1=32,
    // tile_dim0=1024, tile_dim1=32, tensor_dim0_stride=1024
    v8i g1 = { (int)0x00010000u, (int)(1024u << 16), (int)(32u << 16),
               (int)(1024u << 16), 32, 1024, 0, 0 };
    v4i g2 = { 0, 0, 0, 0 };
    v4i g3 = { 0, 0, 0, 0 };
#if __clang_major__ >= 23
    v8i g4 = { 0, 0, 0, 0, 0, 0, 0, 0 };
    __builtin_amdgcn_tensor_load_to_lds(g0, g1, g2, g3, g4, 0);
#else
    __builtin_amdgcn_tensor_load_to_lds(g0, g1, g2, g3, 0);
#endif
#if __has_builtin(__builtin_amdgcn_s_wait_tensorcnt)
    __builtin_amdgcn_s_wait_tensorcnt(0);
#else
    asm volatile("s_wait_tensorcnt 0x0" ::: "memory");
#endif
  }
#else
  {
    const int4* src = (const int4*)(qrow + (size_t)q0 * C_);
    int4* dst = (int4*)&s_q[0][0];
#pragma unroll
    for (int i = 0; i < 16; ++i) dst[threadIdx.x + 256 * i] = src[threadIdx.x + 256 * i];
  }
#endif

  v8f acc[2][8];
#pragma unroll
  for (int r = 0; r < 2; ++r)
#pragma unroll
    for (int i = 0; i < 8; ++i) acc[r][i] = v8f{};

  if (threadIdx.x < 32) { s_m[threadIdx.x] = -__builtin_inff(); s_l[threadIdx.x] = 0.f; }
  __syncthreads();

  int col = lane & 15, hi = lane >> 4;
  int rrow = threadIdx.x >> 3;
  int rj   = threadIdx.x & 7;
  int s_end = q0 + 32;

  for (int kb0 = 0; kb0 < s_end; kb0 += 128) {
    int rem  = s_end - kb0;
    int kcap = rem >= 128 ? 128 : ((rem + 31) & ~31);
    int s0 = kb0 + 16 * wave;

    // prefetch this key block's Vt tile (covered by the score GEMM below)
    {
      const __bf16* pv = vrow + (size_t)(128 * wave + lane * 4) * T_ + kb0;
      __builtin_prefetch(pv,          0, 1);
      __builtin_prefetch(pv + T_,     0, 1);
      __builtin_prefetch(pv + 2 * T_, 0, 1);
      __builtin_prefetch(pv + 3 * T_, 0, 1);
    }

    // ---- scores: 32x16 tile per wave, parity ping-pong on k-fragments ----
    if (16 * wave < rem) {
      v8f sc0 = v8f{}, sc1 = v8f{};
      v16bf kf[2];
      kf[0] = load_frag_b(krow, C_, s0, 0, lane);
#pragma unroll
      for (int st = 0; st < 32; ++st) {
        if (st < 31) kf[(st + 1) & 1] = load_frag_b(krow, C_, s0, 32 * (st + 1), lane);
        v16bf aA = load_frag_a(&s_q[0][0], C_, 0,  32 * st, lane);
        v16bf aB = load_frag_a(&s_q[0][0], C_, 16, 32 * st, lane);
        sc0 = wmma_bf16(aA, kf[st & 1], sc0);
        sc1 = wmma_bf16(aB, kf[st & 1], sc1);
      }
#pragma unroll
      for (int j = 0; j < 8; ++j) {
        int qrA = j + 8 * hi;
        int qrB = 16 + j + 8 * hi;
        int s  = s0 + col;
        float vA = sc0[j], vB = sc1[j];
        if (s > q0 + qrA) vA = -__builtin_inff();
        if (s > q0 + qrB) vB = -__builtin_inff();
        s_s[qrA][16 * wave + col] = vA;
        s_s[qrB][16 * wave + col] = vB;
      }
    } else if (16 * wave < kcap) {
#pragma unroll
      for (int j = 0; j < 8; ++j) {
        s_s[j + 8 * hi][16 * wave + col]      = -__builtin_inff();
        s_s[16 + j + 8 * hi][16 * wave + col] = -__builtin_inff();
      }
    }
    __syncthreads();                                             // B1

    // ---- parallel online softmax (32 rows x 8 slots) ----
    float pm = -__builtin_inff();
    for (int i = rj; i < kcap; i += 8) pm = fmaxf(pm, s_s[rrow][i]);
    s_red[rrow][rj] = pm;
    __syncthreads();                                             // B2
    if (rj == 0) {
      float mb = s_red[rrow][0];
#pragma unroll
      for (int i = 1; i < 8; ++i) mb = fmaxf(mb, s_red[rrow][i]);
      float mn = fmaxf(s_m[rrow], mb);
      s_r[rrow] = __expf(s_m[rrow] - mn);
      s_m[rrow] = mn;
    }
    __syncthreads();                                             // B3
    {
      float mn = s_m[rrow];
      float psum = 0.f;
      for (int i = rj; i < kcap; i += 8) {
        float p = __expf(s_s[rrow][i] - mn);
        s_p[rrow][i] = (__bf16)p;
        psum += p;
      }
      s_red[rrow][rj] = psum;
    }
    __syncthreads();                                             // B4
    if (rj == 0) {
      float sum = 0.f;
#pragma unroll
      for (int i = 0; i < 8; ++i) sum += s_red[rrow][i];
      s_l[rrow] = s_l[rrow] * s_r[rrow] + sum;
    }

    // ---- rescale accumulators, then P x Vt (parity ping-pong B) ----
#pragma unroll
    for (int r = 0; r < 2; ++r)
#pragma unroll
      for (int i = 0; i < 8; ++i)
#pragma unroll
        for (int j = 0; j < 8; ++j) acc[r][i][j] *= s_r[16 * r + j + 8 * hi];

    for (int ks = 0; ks < kcap; ks += 32) {
      v16bf ap0 = load_frag_a(&s_p[0][0], 128, 0,  ks, lane);
      v16bf ap1 = load_frag_a(&s_p[0][0], 128, 16, ks, lane);
      v16bf bb[2];
      bb[0] = load_frag_b(vrow, T_, 128 * wave, kb0 + ks, lane);
#pragma unroll
      for (int i = 0; i < 8; ++i) {
        if (i < 7) bb[(i + 1) & 1] = load_frag_b(vrow, T_, 128 * wave + 16 * (i + 1), kb0 + ks, lane);
        acc[0][i] = wmma_bf16(ap0, bb[i & 1], acc[0][i]);
        acc[1][i] = wmma_bf16(ap1, bb[i & 1], acc[1][i]);
      }
    }
    __syncthreads();                                             // B5
  }

  // ---- epilogue: y = acc / l ----
#pragma unroll
  for (int r = 0; r < 2; ++r) {
#pragma unroll
    for (int i = 0; i < 8; ++i) {
      int c = 128 * wave + 16 * i + col;
#pragma unroll
      for (int j = 0; j < 8; ++j) {
        int qr = 16 * r + j + 8 * hi;
        out[((size_t)bidx * T_ + q0 + qr) * C_ + c] = acc[r][i][j] / s_l[qr];
      }
    }
  }
}

// ---------------------------------------------------------------------------
extern "C" void kernel_launch(void* const* d_in, const int* in_sizes, int n_in,
                              void* d_out, int out_size, void* d_ws, size_t ws_size,
                              hipStream_t stream) {
  const float* x = (const float*)d_in[0];   // [B,T,C]
  const float* w = (const float*)d_in[1];   // [3C,C]
  float* out = (float*)d_out;               // [B,T,C]

  __bf16* xb = (__bf16*)d_ws;
  __bf16* wb = xb + (size_t)B_ * T_ * C_;
  __bf16* qb = wb + (size_t)NQKV * C_;
  __bf16* kb = qb + (size_t)B_ * T_ * C_;
  __bf16* vt = kb + (size_t)B_ * T_ * C_;

  cvt_bf16<<<2048, 256, 0, stream>>>(x, xb, (size_t)B_ * T_ * C_ / 4);
  cvt_bf16<<<1024, 256, 0, stream>>>(w, wb, (size_t)NQKV * C_ / 4);

  dim3 g1((B_ * T_) / 256, NQKV / 64);
  qkv_gemm<<<g1, 256, 0, stream>>>(xb, wb, qb, kb, vt);

  dim3 g2(T_ / 32, B_);
  flash_attn<<<g2, 256, 0, stream>>>(qb, kb, vt, out);
}
